// EncoderLayer_hybrid1_8315056685231
// MI455X (gfx1250) — compile-verified, matrix operand order checked by reading, not run on hardware
//
#include <hip/hip_runtime.h>
#include <hip/hip_bf16.h>
#include <cmath>
#include <cstdint>

typedef __attribute__((ext_vector_type(16))) __bf16 v16bf;
typedef __attribute__((ext_vector_type(8)))  float  v8f;
typedef int vi4 __attribute__((vector_size(16)));  // matches builtin pointee

namespace {
constexpr int Bb = 16, Nn = 1024, Ee = 512, Hh = 8, DH = 64;
constexpr float EPS = 1e-5f;
}

// ---- feature guards --------------------------------------------------------
#if defined(__has_builtin)
#  if __has_builtin(__builtin_amdgcn_global_load_async_to_lds_b128) && \
      __has_builtin(__builtin_amdgcn_s_wait_asynccnt)
#    define ASYNC_LDS 1
#  endif
#  if __has_builtin(__builtin_amdgcn_permlanex16)
#    define HAS_PERMX16 1
#  endif
#endif
#ifndef ASYNC_LDS
#define ASYNC_LDS 0
#endif
#ifndef HAS_PERMX16
#define HAS_PERMX16 0
#endif

#if ASYNC_LDS
// 16B global -> LDS async copy (ASYNCcnt-tracked). Builtin signature is
// void(v4i AS1*, v4i AS3*, imm offset, imm cpol). AS1 shares the flat
// representation for global memory, so an integer round-trip is safe there
// (and drops const); the LDS side is a real generic->AS3 addrspacecast.
#define GLD_ASYNC_B128(gsrc, ldst)                                         \
  __builtin_amdgcn_global_load_async_to_lds_b128(                          \
      (__attribute__((address_space(1))) vi4*)(uintptr_t)(const void*)(gsrc), \
      (__attribute__((address_space(3))) vi4*)(ldst), 0, 0)
#endif

// ---- WMMA helpers (CDNA5 wave32, 16x16x32 bf16 -> f32) ---------------------
__device__ __forceinline__ int kmap16(int j, int half) {
  // 16-bit A/B fragment K index: lanes 0-15 hold K {0..7,16..23}, lanes 16-31
  // hold K {8..15,24..31} (ISA 7.12.2).
  return ((j < 8) ? j : (j + 8)) + (half << 3);
}

__device__ __forceinline__ v8f wmma_bf16(v16bf a, v16bf b, v8f c) {
  return __builtin_amdgcn_wmma_f32_16x16x32_bf16(
      /*neg_a=*/false, a, /*neg_b=*/false, b,
      /*c_mod=*/(short)0, c, /*reuse_a=*/false, /*reuse_b=*/false);
}

// ---- DPP16 butterfly reductions (no LDS, no dscnt waits) -------------------
// quad_perm xor1 = 0xB1, xor2 = 0x4E; row_ror:4 = 0x124, row_ror:8 = 0x128.
// All stay within a 16-lane row; wave fully active so bound_ctrl is moot.
#define DPP_F32(v, ctrl)                                                     \
  __int_as_float(__builtin_amdgcn_update_dpp(0, __float_as_int(v), (ctrl),   \
                                             0xF, 0xF, true))

__device__ __forceinline__ float redmax16(float v) {
  v = fmaxf(v, DPP_F32(v, 0xB1));
  v = fmaxf(v, DPP_F32(v, 0x4E));
  v = fmaxf(v, DPP_F32(v, 0x124));
  v = fmaxf(v, DPP_F32(v, 0x128));
  return v;
}
__device__ __forceinline__ float redsum16(float v) {
  v += DPP_F32(v, 0xB1);
  v += DPP_F32(v, 0x4E);
  v += DPP_F32(v, 0x124);
  v += DPP_F32(v, 0x128);
  return v;
}
__device__ __forceinline__ float redsum32(float v) {
  v = redsum16(v);
#if HAS_PERMX16
  v += __int_as_float(__builtin_amdgcn_permlanex16(
      __float_as_int(v), __float_as_int(v), 0x76543210, (int)0xfedcba98,
      false, false));
#else
  v += __shfl_xor(v, 16, 32);
#endif
  return v;
}

// ---- 1) Build per-head orthogonal W(phi): 125 Givens rotations -------------
// Thread t owns column t; rotations mix rows only -> no cross-thread deps.
__global__ __launch_bounds__(64) void build_worth_k(const float* __restrict__ phi,
                                                    float* __restrict__ worth) {
  __shared__ float W[DH][DH];
  const int h = blockIdx.x, t = threadIdx.x;
  for (int r = 0; r < DH; ++r) W[r][t] = (r == t) ? 1.0f : 0.0f;
  const float* ph = phi + h * (2 * DH - 3);
  for (int g = 0; g < 2 * DH - 3; ++g) {
    int k = (g < DH - 1) ? g : (2 * DH - 4 - g);   // 0..62 then 61..0
    float c = cosf(ph[g]), s = sinf(ph[g]);
    float rk = W[k][t], rk1 = W[k + 1][t];
    W[k][t]     =  c * rk + s * rk1;
    W[k + 1][t] = -s * rk + c * rk1;
  }
  for (int r = 0; r < DH; ++r)
    worth[(size_t)h * DH * DH + r * DH + t] = W[r][t];
}

// ---- 2) fp32 -> bf16 weight conversion -------------------------------------
__global__ void cvt_w_k(const float* __restrict__ W1, const float* __restrict__ W2,
                        __bf16* __restrict__ w1b, __bf16* __restrict__ w2b) {
  int i = blockIdx.x * 256 + threadIdx.x;
  w1b[i] = (__bf16)W1[i];
  w2b[i] = (__bf16)W2[i];
}

// ---- 3) LN1 + per-head Q = xh@Worth, K = xh, V = xh@Wv + bv ----------------
// One wave per token row. q/k/v written bf16 in head-major [B*H][N][dh].
__global__ __launch_bounds__(256) void ln1_qkv_k(
    const float* __restrict__ x, const float* __restrict__ worth,
    const float* __restrict__ Wv, const float* __restrict__ bv,
    float* __restrict__ xn, __bf16* __restrict__ qg,
    __bf16* __restrict__ kg, __bf16* __restrict__ vg) {
  __shared__ float xs[8][Ee];
  const int tid = threadIdx.x, lane = tid & 31, wv = tid >> 5;
  const size_t row = (size_t)blockIdx.x * 8 + wv;
  const float* xr = x + row * Ee;
  float vals[16], s = 0.f, ss = 0.f;
#pragma unroll
  for (int i = 0; i < 16; ++i) {
    float f = xr[lane + i * 32];
    vals[i] = f; s += f; ss += f * f;
  }
  s = redsum32(s); ss = redsum32(ss);
  float mean = s * (1.0f / Ee);
  float var  = ss * (1.0f / Ee) - mean * mean;
  float rinv = rsqrtf(var + EPS);
  const int b = (int)(row >> 10), n = (int)(row & 1023);
#pragma unroll
  for (int i = 0; i < 16; ++i) {
    int idx = lane + i * 32;
    float f = (vals[i] - mean) * rinv;
    xs[wv][idx] = f;
    xn[row * Ee + idx] = f;
    int h = idx >> 6, d = idx & 63;
    kg[((size_t)(b * Hh + h) * Nn + n) * DH + d] = (__bf16)f;
  }
  // same-wave LDS ordering guarantees xs visibility; no barrier needed
  for (int h = 0; h < Hh; ++h) {
    const float* wo  = worth + (size_t)h * DH * DH;
    const float* wvh = Wv + (size_t)h * DH * DH;
    const int e0 = lane * 2;
    float q0 = 0.f, q1 = 0.f, v0 = 0.f, v1 = 0.f;
    for (int d = 0; d < DH; ++d) {
      float xk = xs[wv][h * DH + d];
      q0 += xk * wo[d * DH + e0];
      q1 += xk * wo[d * DH + e0 + 1];
      v0 += xk * wvh[d * DH + e0];
      v1 += xk * wvh[d * DH + e0 + 1];
    }
    v0 += bv[h * DH + e0]; v1 += bv[h * DH + e0 + 1];
    size_t base = ((size_t)(b * Hh + h) * Nn + n) * DH + e0;
    qg[base] = (__bf16)q0; qg[base + 1] = (__bf16)q1;
    vg[base] = (__bf16)v0; vg[base + 1] = (__bf16)v1;
  }
}

// ---- 4) Flash attention: P = softmax(Q K^T / 8), O = P V -------------------
// Block = one (b,h) and 128 query rows; 8 waves x 16 rows; key blocks of 32.
__global__ __launch_bounds__(256) void flash_attn_k(
    const __bf16* __restrict__ q, const __bf16* __restrict__ kk,
    const __bf16* __restrict__ vv, float* __restrict__ attn) {
  __shared__ __attribute__((aligned(16))) __bf16 Kt[32 * DH];
  __shared__ __attribute__((aligned(16))) __bf16 Vt[32 * DH];
  __shared__ __attribute__((aligned(16))) __bf16 Pt[8][16 * 32];
  const int tid = threadIdx.x, lane = tid & 31, wv = tid >> 5;
  const int half = lane >> 4, l16 = lane & 15;
  const int bh = blockIdx.y;
  const int q0 = blockIdx.x * 128 + wv * 16;
  const __bf16* qb = q  + (size_t)bh * Nn * DH;
  const __bf16* kb = kk + (size_t)bh * Nn * DH;
  const __bf16* vb = vv + (size_t)bh * Nn * DH;

  // Q fragments (A-layout), held in registers for the whole kernel
  v16bf aq0, aq1;
#pragma unroll
  for (int j = 0; j < 16; ++j) {
    int km = kmap16(j, half);
    aq0[j] = qb[(size_t)(q0 + l16) * DH + km];
    aq1[j] = qb[(size_t)(q0 + l16) * DH + 32 + km];
  }

  v8f O[4] = {};
  float m[8], l[8];
#pragma unroll
  for (int r = 0; r < 8; ++r) { m[r] = -1e30f; l[r] = 0.0f; }
  const float scale = 0.125f;  // 1/sqrt(dh)

  for (int kblk = 0; kblk < Nn / 32; ++kblk) {
    __syncthreads();
    // K/V tiles are contiguous 4KB blocks in head-major layout: 16B/thread
#if ASYNC_LDS
    GLD_ASYNC_B128(kb + (size_t)kblk * 32 * DH + tid * 8, Kt + tid * 8);
    GLD_ASYNC_B128(vb + (size_t)kblk * 32 * DH + tid * 8, Vt + tid * 8);
    __builtin_amdgcn_s_wait_asynccnt(0);
#else
    ((uint4*)Kt)[tid] = ((const uint4*)(kb + (size_t)kblk * 32 * DH))[tid];
    ((uint4*)Vt)[tid] = ((const uint4*)(vb + (size_t)kblk * 32 * DH))[tid];
#endif
    if (kblk + 1 < Nn / 32) {
      __builtin_prefetch(kb + (size_t)(kblk + 1) * 32 * DH + tid * 8, 0, 0);
      __builtin_prefetch(vb + (size_t)(kblk + 1) * 32 * DH + tid * 8, 0, 0);
    }
    __syncthreads();

    // S(16x32) = Q(16x64) . K_blk^T : 2 n-tiles x 2 K-chunks = 4 WMMA
    v16bf b0, b1;
    v8f c0 = {}, c1 = {};
#pragma unroll
    for (int j = 0; j < 16; ++j) {
      int km = kmap16(j, half);
      b0[j] = Kt[l16 * DH + km];
      b1[j] = Kt[(16 + l16) * DH + km];
    }
    c0 = wmma_bf16(aq0, b0, c0);
    c1 = wmma_bf16(aq0, b1, c1);
#pragma unroll
    for (int j = 0; j < 16; ++j) {
      int km = kmap16(j, half);
      b0[j] = Kt[l16 * DH + 32 + km];
      b1[j] = Kt[(16 + l16) * DH + 32 + km];
    }
    c0 = wmma_bf16(aq1, b0, c0);
    c1 = wmma_bf16(aq1, b1, c1);

    // Online softmax; C-frag row = r + 8*half, cols across lanes 0..15.
    // Row stats via DPP16 butterflies (pure VALU, no DS traffic).
#pragma unroll
    for (int r = 0; r < 8; ++r) {
      float s0 = c0[r] * scale, s1 = c1[r] * scale;
      float rm = redmax16(fmaxf(s0, s1));
      float mn = fmaxf(m[r], rm);
      float alpha = __expf(m[r] - mn);
      float p0 = __expf(s0 - mn), p1 = __expf(s1 - mn);
      float rs = redsum16(p0 + p1);
      l[r] = l[r] * alpha + rs;
      m[r] = mn;
#pragma unroll
      for (int e = 0; e < 4; ++e) O[e][r] *= alpha;
      int prow = r + 8 * half;
      Pt[wv][prow * 32 + l16]      = (__bf16)p0;
      Pt[wv][prow * 32 + 16 + l16] = (__bf16)p1;
    }

    // O += P(16x32) . V_blk(32x64): 4 WMMA (per-wave LDS, wave-ordered DS)
    v16bf ap;
#pragma unroll
    for (int j = 0; j < 16; ++j) ap[j] = Pt[wv][l16 * 32 + kmap16(j, half)];
#pragma unroll
    for (int e = 0; e < 4; ++e) {
      v16bf bvf;
#pragma unroll
      for (int j = 0; j < 16; ++j)
        bvf[j] = Vt[kmap16(j, half) * DH + e * 16 + l16];
      O[e] = wmma_bf16(ap, bvf, O[e]);
    }
  }

#pragma unroll
  for (int r = 0; r < 8; ++r) {
    float inv = 1.0f / l[r];
    int row = q0 + r + 8 * half;
#pragma unroll
    for (int e = 0; e < 4; ++e)
      attn[((size_t)bh * Nn + row) * DH + e * 16 + l16] = O[e][r] * inv;
  }
}

// ---- 5) head_out = LN_dh(attn + xh); res = head_out + x --------------------
__global__ __launch_bounds__(256) void headln_res_k(
    const float* __restrict__ attn, const float* __restrict__ xn,
    const float* __restrict__ x, float* __restrict__ res) {
  const int tid = threadIdx.x, lane = tid & 31, wv = tid >> 5;
  const size_t u = (size_t)blockIdx.x * 8 + wv;  // ((b*N+n)*H + h)
  const int h = (int)(u & 7);
  const size_t bn = u >> 3;
  const int n = (int)(bn & 1023), b = (int)(bn >> 10);
  const size_t aidx = (((size_t)(b * Hh + h) * Nn) + n) * DH + lane * 2;
  const size_t xidx = ((size_t)b * Nn + n) * Ee + h * DH + lane * 2;
  float y0 = attn[aidx] + xn[xidx];
  float y1 = attn[aidx + 1] + xn[xidx + 1];
  float s  = redsum32(y0 + y1);
  float ss = redsum32(y0 * y0 + y1 * y1);
  float mean = s * (1.0f / DH);
  float var  = ss * (1.0f / DH) - mean * mean;
  float rinv = rsqrtf(var + EPS);
  res[xidx]     = (y0 - mean) * rinv + x[xidx];
  res[xidx + 1] = (y1 - mean) * rinv + x[xidx + 1];
}

// ---- 6) t2 = bf16(LN_E(res)) -----------------------------------------------
__global__ __launch_bounds__(256) void ln2_k(const float* __restrict__ res,
                                             __bf16* __restrict__ t2) {
  const int tid = threadIdx.x, lane = tid & 31, wv = tid >> 5;
  const size_t row = (size_t)blockIdx.x * 8 + wv;
  const float* rr = res + row * Ee;
  float vals[16], s = 0.f, ss = 0.f;
#pragma unroll
  for (int i = 0; i < 16; ++i) {
    float f = rr[lane + i * 32];
    vals[i] = f; s += f; ss += f * f;
  }
  s = redsum32(s); ss = redsum32(ss);
  float mean = s * (1.0f / Ee), var = ss * (1.0f / Ee) - mean * mean;
  float rinv = rsqrtf(var + EPS);
#pragma unroll
  for (int i = 0; i < 16; ++i)
    t2[row * Ee + lane + i * 32] = (__bf16)((vals[i] - mean) * rinv);
}

// ---- 7) WMMA GEMM: out = act(A[16384x512] @ W[512x512] + b) (+resid) -------
// mode 0: GELU(exact) -> bf16 out.  mode 1: + resid -> fp32 out.
__global__ __launch_bounds__(256) void gemm512_k(
    const __bf16* __restrict__ A, const __bf16* __restrict__ W,
    const float* __restrict__ bias, const float* __restrict__ resid,
    void* __restrict__ outp, int mode) {
  __shared__ __attribute__((aligned(16))) __bf16 At[128 * 32];
  __shared__ __attribute__((aligned(16))) __bf16 Bt[32 * 64];
  const int tid = threadIdx.x, lane = tid & 31, wv = tid >> 5;
  const int half = lane >> 4, l16 = lane & 15;
  const int n0 = blockIdx.x * 64;
  const int m0 = blockIdx.y * 128;
  v8f acc[4] = {};
  for (int kc = 0; kc < Ee / 32; ++kc) {
    __syncthreads();
#if ASYNC_LDS
    {  // A tile 128x32 (8KB) + W tile 32x64 (4KB), all 16B-aligned
      int row = tid >> 1, c0 = (tid & 1) * 16;
      const __bf16* s0 = A + (size_t)(m0 + row) * Ee + kc * 32 + c0;
      GLD_ASYNC_B128(s0,     At + row * 32 + c0);
      GLD_ASYNC_B128(s0 + 8, At + row * 32 + c0 + 8);
      int kr = tid >> 3, c1 = (tid & 7) * 8;
      GLD_ASYNC_B128(W + (size_t)(kc * 32 + kr) * Ee + n0 + c1,
                     Bt + kr * 64 + c1);
      __builtin_amdgcn_s_wait_asynccnt(0);
    }
#else
    {
      int row = tid >> 1, c0 = (tid & 1) * 16;
      const uint4* src = (const uint4*)(A + (size_t)(m0 + row) * Ee + kc * 32 + c0);
      uint4* dst = (uint4*)(At + row * 32 + c0);
      dst[0] = src[0]; dst[1] = src[1];
    }
    {
      int kr = tid >> 3, c1 = (tid & 7) * 8;
      *(uint4*)(Bt + kr * 64 + c1) =
          *(const uint4*)(W + (size_t)(kc * 32 + kr) * Ee + n0 + c1);
    }
#endif
    __syncthreads();
    v16bf a;
#pragma unroll
    for (int j = 0; j < 16; ++j)
      a[j] = At[(wv * 16 + l16) * 32 + kmap16(j, half)];
#pragma unroll
    for (int e = 0; e < 4; ++e) {
      v16bf b;
#pragma unroll
      for (int j = 0; j < 16; ++j)
        b[j] = Bt[kmap16(j, half) * 64 + e * 16 + l16];
      acc[e] = wmma_bf16(a, b, acc[e]);
    }
  }
#pragma unroll
  for (int e = 0; e < 4; ++e) {
#pragma unroll
    for (int r = 0; r < 8; ++r) {
      int row = m0 + wv * 16 + r + 8 * half;
      int col = n0 + e * 16 + l16;
      float val = acc[e][r] + bias[col];
      if (mode == 0) {
        val = 0.5f * val * (1.0f + erff(val * 0.70710678118654752f));
        ((__bf16*)outp)[(size_t)row * Ee + col] = (__bf16)val;
      } else {
        ((float*)outp)[(size_t)row * Ee + col] =
            val + resid[(size_t)row * Ee + col];
      }
    }
  }
}

// ---- launch ----------------------------------------------------------------
extern "C" void kernel_launch(void* const* d_in, const int* in_sizes, int n_in,
                              void* d_out, int out_size, void* d_ws, size_t ws_size,
                              hipStream_t stream) {
  const float* x   = (const float*)d_in[0];
  const float* Wv  = (const float*)d_in[1];
  const float* bv  = (const float*)d_in[2];
  const float* phi = (const float*)d_in[3];
  const float* W1  = (const float*)d_in[4];
  const float* b1  = (const float*)d_in[5];
  const float* W2  = (const float*)d_in[6];
  const float* b2  = (const float*)d_in[7];
  float* out = (float*)d_out;

  char* ws = (char*)d_ws;
  size_t off = 0;
  auto alloc = [&](size_t bytes) -> char* {
    char* p = ws + off;
    off += (bytes + 255) & ~(size_t)255;
    return p;
  };
  const size_t T = (size_t)Bb * Nn;  // 16384 token rows
  float*  worth = (float*)alloc((size_t)Hh * DH * DH * 4);
  float*  xn    = (float*)alloc(T * Ee * 4);
  __bf16* qg    = (__bf16*)alloc(T * Ee * 2);
  __bf16* kg    = (__bf16*)alloc(T * Ee * 2);
  __bf16* vg    = (__bf16*)alloc(T * Ee * 2);
  float*  attn  = (float*)alloc(T * Ee * 4);
  float*  res   = (float*)alloc(T * Ee * 4);
  __bf16* w1b   = (__bf16*)alloc((size_t)Ee * Ee * 2);
  __bf16* w2b   = (__bf16*)alloc((size_t)Ee * Ee * 2);
  __bf16* t2 = qg;  // reuse: qg dead after flash_attn
  __bf16* h1 = kg;  // reuse: kg dead after flash_attn

  build_worth_k<<<Hh, 64, 0, stream>>>(phi, worth);
  cvt_w_k<<<(Ee * Ee) / 256, 256, 0, stream>>>(W1, W2, w1b, w2b);
  ln1_qkv_k<<<(unsigned)(T / 8), 256, 0, stream>>>(x, worth, Wv, bv, xn, qg, kg, vg);
  flash_attn_k<<<dim3(Nn / 128, Bb * Hh), 256, 0, stream>>>(qg, kg, vg, attn);
  headln_res_k<<<(unsigned)((T * Hh) / 8), 256, 0, stream>>>(attn, xn, x, res);
  ln2_k<<<(unsigned)(T / 8), 256, 0, stream>>>(res, t2);
  gemm512_k<<<dim3(Ee / 64, (unsigned)(T / 128)), 256, 0, stream>>>(
      t2, w1b, b1, nullptr, (void*)h1, 0);
  gemm512_k<<<dim3(Ee / 64, (unsigned)(T / 128)), 256, 0, stream>>>(
      h1, w2b, b2, res, (void*)out, 1);
  (void)in_sizes; (void)n_in; (void)out_size; (void)ws_size;
}
